// BackflowNet_48627619726080
// MI455X (gfx1250) — matrix operand order repaired and checked by reading, not compiled, not run on hardware
//
#include <hip/hip_runtime.h>
#include <math.h>

// ---------------------------------------------------------------------------
// BackflowNet fused kernel for MI455X (gfx1250, wave32, WMMA).
// One workgroup (256 thr = 8 waves) per batch element. All intermediates in
// LDS. The dominant [4096,64]x[64,64] e2v GEMM per layer runs on
// v_wmma_f32_16x16x32_f16 with f32 accumulation, chained over source nodes
// so the i-reduction happens inside the WMMA C operand.
// ---------------------------------------------------------------------------

typedef __attribute__((ext_vector_type(16))) _Float16 v16h;
typedef __attribute__((ext_vector_type(8)))  _Float16 v8h;
typedef __attribute__((ext_vector_type(2)))  _Float16 v2h;
typedef __attribute__((ext_vector_type(8)))  float    v8f;

#define NN      64          // nodes per batch
#define HH      64          // hidden width
#define GS_STR  72          // padded f16 row stride (bank-conflict relief, 16B aligned)
#define WT_STR  72
#define SWAPX16 0x401F      // ds_swizzle: group-of-32, xor_mask=0x10, and_mask=0x1f

__device__ __forceinline__ float gelu_f(float v) {
  // exact GELU: x * 0.5 * (1 + erf(x / sqrt(2)))
  return 0.5f * v * (1.0f + erff(v * 0.70710678118654752440f));
}

// Compiler-level fence: intra-wave LDS ops are architecturally in-order
// (DScnt); we only need to stop the compiler from reordering across the
// cross-lane producer/consumer edge.
#define WAVE_SYNC() do { asm volatile("" ::: "memory"); \
                         __builtin_amdgcn_wave_barrier(); \
                         asm volatile("" ::: "memory"); } while (0)

// A fragment (16x32 f16, M=row of tile): lane<16 -> M=lane, K kb+{0..7,16..23};
// lane>=16 -> M=lane-16, K kb+{8..15,24..31}. Two ds_load_b128 per lane.
__device__ __forceinline__ v16h load_a_frag(const _Float16* g, int kb, int lane) {
  const int half = lane >> 4;
  const int r    = lane & 15;
  const _Float16* p = g + r * GS_STR + kb + 8 * half;
  union { v16h v; v8h h[2]; } u;
  u.h[0] = *(const v8h*)(p);
  u.h[1] = *(const v8h*)(p + 16);
  return u.v;
}

// B fragment (32x16 f16, N=col): lane<16 -> N=lane, K kb+{0..7,16..23};
// lane>=16 -> N=lane-16, K kb+{8..15,24..31}. Weights pre-transposed: wt[n][k].
__device__ __forceinline__ v16h load_b_frag(const _Float16* wt, int nb, int kb, int lane) {
  const int half = lane >> 4;
  const int n    = (lane & 15) + nb * 16;
  const _Float16* p = wt + n * WT_STR + kb + 8 * half;
  union { v16h v; v8h h[2]; } u;
  u.h[0] = *(const v8h*)(p);
  u.h[1] = *(const v8h*)(p + 16);
  return u.v;
}

__global__ __launch_bounds__(256) void backflow_fused(
    const float* __restrict__ x,       const float* __restrict__ spin,
    const float* __restrict__ node_w1, const float* __restrict__ node_b1,
    const float* __restrict__ node_w2, const float* __restrict__ node_b2,
    const float* __restrict__ edge_w1, const float* __restrict__ edge_b1,
    const float* __restrict__ edge_w2, const float* __restrict__ edge_b2,
    const float* __restrict__ v2e_w1,  const float* __restrict__ v2e_b1,
    const float* __restrict__ v2e_w2,  const float* __restrict__ v2e_b2,
    const float* __restrict__ e2v_w1,  const float* __restrict__ e2v_b1,
    const float* __restrict__ e2v_w2,  const float* __restrict__ e2v_b2,
    const float* __restrict__ head_w1, const float* __restrict__ head_b1,
    const float* __restrict__ head_w2, const float* __restrict__ head_b2,
    const float* __restrict__ scale,   float* __restrict__ out)
{
  const int b    = blockIdx.x;
  const int tid  = threadIdx.x;
  const int lane = tid & 31;
  const int wave = tid >> 5;

  __shared__ alignas(16) float    hv_s[NN][HH];          // node embeddings   16 KB
  __shared__ alignas(16) float    he_s[NN * NN][2];      // edge features     32 KB
  __shared__ alignas(16) float    acc_s[NN][HH];         // aggregation acc   16 KB
  __shared__ alignas(16) _Float16 g_s[8][16][GS_STR];    // per-wave A stage  18 KB
  __shared__ alignas(16) _Float16 w2t_s[HH][WT_STR];     // e2v_w2^T (f16)     9 KB
  // per-layer small weights, staged so inner loops never touch global memory
  __shared__ alignas(16) float vw1_s[132];               // v2e_w1 [66][2]
  __shared__ alignas(16) float ew1_s[128];               // e2v_w1 [2][64]
  __shared__ alignas(16) float eb1_s[64];                // e2v_b1
  __shared__ float vb1_s[2], vw2_s[4], vb2_s[2];
  __shared__ float xs_s[NN][2];
  __shared__ float sp_s[NN];
  __shared__ float dx_s[NN][2];
  __shared__ float mean_s[2];

  // ---- load positions (OMEGA = 1.0 -> sqrt(OMEGA) = 1) ----
  if (tid < NN) {
    xs_s[tid][0] = x[(b * NN + tid) * 2 + 0];
    xs_s[tid][1] = x[(b * NN + tid) * 2 + 1];
    sp_s[tid]    = spin[b * NN + tid];
  }
  __syncthreads();

  // ---- node MLP: hv = gelu([xs,spin]@w1+b1)@w2+b2 (tiny; VALU) ----
  for (int e = tid; e < NN * HH; e += 256) {   // hidden, reuse acc_s
    int j = e >> 6, h = e & 63;
    float v = xs_s[j][0] * node_w1[0 * HH + h]
            + xs_s[j][1] * node_w1[1 * HH + h]
            + sp_s[j]    * node_w1[2 * HH + h] + node_b1[h];
    acc_s[j][h] = gelu_f(v);
  }
  __syncthreads();
  for (int e = tid; e < NN * HH; e += 256) {
    int j = e >> 6, h = e & 63;
    float v = node_b2[h];
    #pragma unroll 8
    for (int k = 0; k < HH; ++k) v += acc_s[j][k] * node_w2[k * HH + h];
    hv_s[j][h] = v;
  }

  // ---- edge init: he[i,j] from (dr, rr, r2) (tiny; VALU) ----
  for (int p = tid; p < NN * NN; p += 256) {
    int i = p >> 6, j = p & 63;
    float d0 = xs_s[j][0] - xs_s[i][0];
    float d1 = xs_s[j][1] - xs_s[i][1];
    float r2 = d0 * d0 + d1 * d1;
    float rr = sqrtf(r2 + 1e-12f);
    float u0 = gelu_f(d0 * edge_w1[0] + d1 * edge_w1[2] + rr * edge_w1[4] + r2 * edge_w1[6] + edge_b1[0]);
    float u1 = gelu_f(d0 * edge_w1[1] + d1 * edge_w1[3] + rr * edge_w1[5] + r2 * edge_w1[7] + edge_b1[1]);
    he_s[p][0] = u0 * edge_w2[0] + u1 * edge_w2[2] + edge_b2[0];
    he_s[p][1] = u0 * edge_w2[1] + u1 * edge_w2[3] + edge_b2[1];
  }
  __syncthreads();

  // ---- message-passing layers ----
  const float inv = 1.0f / (float)(NN - 1);
  for (int l = 0; l < 2; ++l) {
    const float* ww2 = e2v_w2 + l * HH * HH;  // [64][64] (k, h)
    const float* eb2 = e2v_b2 + l * HH;

    // stage e2v_w2^T as f16, small weights, and zero the accumulator
    for (int e = tid; e < NN * HH; e += 256) {
      int n = e >> 6, k = e & 63;
      w2t_s[n][k] = (_Float16)ww2[k * HH + n];
      acc_s[n][k] = 0.0f;
    }
    if (tid < 132) vw1_s[tid] = v2e_w1[l * 132 + tid];
    if (tid < 128) ew1_s[tid] = e2v_w1[l * 128 + tid];
    if (tid < 64)  eb1_s[tid] = e2v_b1[l * 64 + tid];
    if (tid < 2)   { vb1_s[tid] = v2e_b1[l * 2 + tid]; vb2_s[tid] = v2e_b2[l * 2 + tid]; }
    if (tid < 4)   vw2_s[tid] = v2e_w2[l * 4 + tid];
    __syncthreads();

    // hoist all 8 B fragments (4 N-tiles x 2 K-steps) out of the tile loops
    v16h Bf[4][2];
    #pragma unroll
    for (int nb = 0; nb < 4; ++nb) {
      Bf[nb][0] = load_b_frag(&w2t_s[0][0], nb, 0,  lane);
      Bf[nb][1] = load_b_frag(&w2t_s[0][0], nb, 32, lane);
    }

    // wave w owns source nodes i in [8w, 8w+8); reduce over them in the
    // WMMA C operand, flush once per destination block.
    for (int jb4 = 0; jb4 < 4; ++jb4) {
      const int jb = jb4 << 4;
      v8f C[4] = {v8f{}, v8f{}, v8f{}, v8f{}};

      for (int ii = 0; ii < 8; ++ii) {
        const int i     = wave * 8 + ii;
        const int pbase = i * NN + jb;

        // (1) v2e edge update: he' = gelu([hv[i],he]@vw1+vb1)@vw2+vb2.
        // All 32 lanes active: lane = (pair r, out-channel o); the two GELU'd
        // hidden units are exchanged between half-waves via ds_swizzle SWAPX16.
        {
          const int o = lane >> 4;               // output channel 0/1
          const int p = pbase + (lane & 15);     // pair
          float a = vb1_s[o];
          #pragma unroll 8
          for (int k = 0; k < HH; ++k) a += hv_s[i][k] * vw1_s[k * 2 + o];
          a += he_s[p][0] * vw1_s[128 + o] + he_s[p][1] * vw1_s[130 + o];
          float u   = gelu_f(a);
          float uot = __int_as_float(
              __builtin_amdgcn_ds_swizzle(__float_as_int(u), SWAPX16));
          float u0 = o ? uot : u;
          float u1 = o ? u : uot;
          he_s[p][o] = u0 * vw2_s[o] + u1 * vw2_s[2 + o] + vb2_s[o];
        }
        WAVE_SYNC();

        // (2) e2v first layer + GELU, staged as f16 A-tile (16 rows x 64 K).
        // Self-edge mask applied here: zero the whole diagonal row so the
        // GEMM contribution of pair (i,i) vanishes (bias handled in epilogue).
        for (int idx = lane; idx < 16 * 32; idx += 32) {
          int r  = idx >> 5;
          int k0 = (idx & 31) << 1;
          int p  = pbase + r;
          float h0 = he_s[p][0], h1 = he_s[p][1];
          float g0 = gelu_f(h0 * ew1_s[k0]     + h1 * ew1_s[HH + k0]     + eb1_s[k0]);
          float g1 = gelu_f(h0 * ew1_s[k0 + 1] + h1 * ew1_s[HH + k0 + 1] + eb1_s[k0 + 1]);
          if (jb + r == i) { g0 = 0.0f; g1 = 0.0f; }   // branchless cndmask
          v2h pk; pk[0] = (_Float16)g0; pk[1] = (_Float16)g1;
          *(v2h*)&g_s[wave][r][k0] = pk;
        }
        WAVE_SYNC();

        // (3) [16 x 64] x [64 x 64] on the matrix pipe, accumulate over i
        v16h A0 = load_a_frag(&g_s[wave][0][0], 0,  lane);
        v16h A1 = load_a_frag(&g_s[wave][0][0], 32, lane);
        #pragma unroll
        for (int nb = 0; nb < 4; ++nb) {
          C[nb] = __builtin_amdgcn_wmma_f32_16x16x32_f16(false, A0, false, Bf[nb][0],
                                                         (short)0, C[nb], false, false);
          C[nb] = __builtin_amdgcn_wmma_f32_16x16x32_f16(false, A1, false, Bf[nb][1],
                                                         (short)0, C[nb], false, false);
        }
      }

      // flush this wave's 8-source partial sums into the shared accumulator
      const int half = lane >> 4;
      const int nlo  = lane & 15;
      #pragma unroll
      for (int nb = 0; nb < 4; ++nb) {
        const int hcol = nb * 16 + nlo;
        #pragma unroll
        for (int v = 0; v < 8; ++v) {            // D rows: m = v + 8*half
          atomicAdd(&acc_s[jb + v + 8 * half][hcol], C[nb][v]);
        }
      }
    }
    __syncthreads();
    // hv <- hv + inv * sum_gemm + b2   (b2 folded: sum_{i!=j} inv*b2 = b2)
    for (int e = tid; e < NN * HH; e += 256) {
      int j = e >> 6, h = e & 63;
      hv_s[j][h] += acc_s[j][h] * inv + eb2[h];
    }
    __syncthreads();
  }

  // ---- head: dx = tanh(hv@w1+b1)@w2+b2, * softplus(scale), centered ----
  for (int e = tid; e < NN * HH; e += 256) {    // reuse acc_s as hidden
    int j = e >> 6, h = e & 63;
    float v = head_b1[h];
    #pragma unroll 8
    for (int k = 0; k < HH; ++k) v += hv_s[j][k] * head_w1[k * HH + h];
    acc_s[j][h] = tanhf(v);
  }
  __syncthreads();
  const float sp = log1pf(expf(scale[0]));      // softplus
  if (tid < NN * 2) {
    int j = tid >> 1, d = tid & 1;
    float v = head_b2[d];
    #pragma unroll 8
    for (int k = 0; k < HH; ++k) v += acc_s[j][k] * head_w2[k * 2 + d];
    dx_s[j][d] = v * sp;
  }
  __syncthreads();
  if (tid < 2) {
    float s = 0.0f;
    for (int j = 0; j < NN; ++j) s += dx_s[j][tid];
    mean_s[tid] = s * (1.0f / (float)NN);
  }
  __syncthreads();
  if (tid < NN * 2) {
    int j = tid >> 1, d = tid & 1;
    out[(b * NN + j) * 2 + d] = dx_s[j][d] - mean_s[d];
  }
}

extern "C" void kernel_launch(void* const* d_in, const int* in_sizes, int n_in,
                              void* d_out, int out_size, void* d_ws, size_t ws_size,
                              hipStream_t stream) {
  (void)n_in; (void)out_size; (void)d_ws; (void)ws_size;
  const int B = in_sizes[0] / (NN * 2);   // 128
  backflow_fused<<<dim3(B), dim3(256), 0, stream>>>(
      (const float*)d_in[0],  (const float*)d_in[1],
      (const float*)d_in[2],  (const float*)d_in[3],
      (const float*)d_in[4],  (const float*)d_in[5],
      (const float*)d_in[6],  (const float*)d_in[7],
      (const float*)d_in[8],  (const float*)d_in[9],
      (const float*)d_in[10], (const float*)d_in[11],
      (const float*)d_in[12], (const float*)d_in[13],
      (const float*)d_in[14], (const float*)d_in[15],
      (const float*)d_in[16], (const float*)d_in[17],
      (const float*)d_in[18], (const float*)d_in[19],
      (const float*)d_in[20], (const float*)d_in[21],
      (const float*)d_in[22], (float*)d_out);
}
